// SpectralNet_20650202759570
// MI455X (gfx1250) — matrix operand order repaired and checked by reading, not compiled, not run on hardware
//
#include <hip/hip_runtime.h>
#include <hip/hip_bf16.h>

typedef __attribute__((ext_vector_type(16))) _Float16 v16h;
typedef __attribute__((ext_vector_type(8)))  float    v8f;

#define N_NODES 50000
#define N_EDGES 800000
#define F_CAT   192
#define KCLU    16
#define M_SUB   256

// ---------------------------------------------------------------------------
// WMMA GEMM, LDS-staged. Block = 256 threads = 8 waves.
// Block computes one 16-row strip x 128 columns (8 wmma subtiles, 1/wave).
// A(m,k) = A[m*ldaR + k*ldaC], B(k,n) = B[k*ldbR + n*ldbC] (strides allow
// transposed views). f32 operands converted to f16 at staging; f32 accumulate.
// Staging has compile-time trip counts (column index clamped, K-tail handled
// branchlessly via clamp+select in the GUARD instantiation).
// atomicMode=1: split-K over blockIdx.y, atomicAdd into pre-zeroed C.
// ---------------------------------------------------------------------------
template<bool GUARD>
__device__ __forceinline__ void stage_tiles(
    const float* __restrict__ A, long long ldaR, long long ldaC,
    const float* __restrict__ B, long long ldbR, long long ldbC,
    _Float16 (*As)[32], _Float16 (*Bs)[32],
    int m0, int nBase, int NdMax, int k0, int kEnd, int tid, int bNFast)
{
  // ---- A tile: 16 rows x 32 k, k-fastest (coalesced when ldaC==1) ----
  float va[2];
#pragma unroll
  for (int i = 0; i < 2; ++i) {
    int idx = tid + i * 256;
    int m = idx >> 5, k = idx & 31;
    int kk = k0 + k;
    int kc = GUARD ? ((kk < kEnd) ? kk : (kEnd - 1)) : kk;
    float v = A[(long long)(m0 + m) * ldaR + (long long)kc * ldaC];
    if (GUARD && kk >= kEnd) v = 0.f;
    va[i] = v;
  }
#pragma unroll
  for (int i = 0; i < 2; ++i) {
    int idx = tid + i * 256;
    As[idx >> 5][idx & 31] = (_Float16)va[i];
  }
  // ---- B tile: 128 cols x 32 k (column clamped -> always in-bounds) ----
  float vb[16];
  if (bNFast) {   // consecutive lanes -> consecutive n (coalesced, ldbC==1)
#pragma unroll
    for (int i = 0; i < 16; ++i) {
      int idx = tid + i * 256;
      int n = idx & 127, k = idx >> 7;
      int nn = nBase + n; nn = (nn > NdMax) ? NdMax : nn;
      int kk = k0 + k;
      int kc = GUARD ? ((kk < kEnd) ? kk : (kEnd - 1)) : kk;
      float v = B[(long long)kc * ldbR + (long long)nn * ldbC];
      if (GUARD && kk >= kEnd) v = 0.f;
      vb[i] = v;
    }
#pragma unroll
    for (int i = 0; i < 16; ++i) {
      int idx = tid + i * 256;
      Bs[idx & 127][idx >> 7] = (_Float16)vb[i];
    }
  } else {        // consecutive lanes -> consecutive k (coalesced, ldbR==1)
#pragma unroll
    for (int i = 0; i < 16; ++i) {
      int idx = tid + i * 256;
      int n = idx >> 5, k = idx & 31;
      int nn = nBase + n; nn = (nn > NdMax) ? NdMax : nn;
      int kk = k0 + k;
      int kc = GUARD ? ((kk < kEnd) ? kk : (kEnd - 1)) : kk;
      float v = B[(long long)kc * ldbR + (long long)nn * ldbC];
      if (GUARD && kk >= kEnd) v = 0.f;
      vb[i] = v;
    }
#pragma unroll
    for (int i = 0; i < 16; ++i) {
      int idx = tid + i * 256;
      Bs[idx >> 5][idx & 31] = (_Float16)vb[i];
    }
  }
}

__device__ __forceinline__ v8f wmma_step(const _Float16 (*As)[32],
                                         const _Float16 (*Bs)[32],
                                         int lane, int w, v8f acc)
{
  const int mr = lane & 15;
  const int aK = (lane & 16) ? 8 : 0;    // ISA 16-bit A 16x32 lane layout
  const int bK = (lane & 16) ? 8 : 0;    // ISA 16-bit B 32x16 lane layout
  const int nL = (w << 4) + mr;
  union Uf { float4 f; _Float16 h[8]; };
  Uf a0, a1, b0, b1;
  a0.f = *(const float4*)&As[mr][aK];        // ds_load_b128
  a1.f = *(const float4*)&As[mr][16 + aK];
  b0.f = *(const float4*)&Bs[nL][bK];
  b1.f = *(const float4*)&Bs[nL][16 + bK];
  v16h af, bf;
#pragma unroll
  for (int i = 0; i < 8; ++i) {
    af[i] = a0.h[i]; af[i + 8] = a1.h[i];
    bf[i] = b0.h[i]; bf[i + 8] = b1.h[i];
  }
  return __builtin_amdgcn_wmma_f32_16x16x32_f16(false, af, false, bf,
                                                (short)0, acc, false, false);
}

__global__ __launch_bounds__(256) void gemm_wmma(
    const float* __restrict__ A, long long ldaR, long long ldaC,
    const float* __restrict__ B, long long ldbR, long long ldbC,
    const float* __restrict__ bias,
    float* __restrict__ C, long long ldc,
    int M, int Nd, int Kd, int kChunk, int relu, int atomicMode, int ntg,
    int bNFast)
{
  __shared__ __align__(16) _Float16 As[16][32];     // 1 KB
  __shared__ __align__(16) _Float16 Bs[128][32];    // 8 KB
  const int tid  = threadIdx.x;
  const int lane = tid & 31;
  const int w    = tid >> 5;
  const int mt    = blockIdx.x / ntg;
  const int g     = blockIdx.x % ntg;
  const int m0    = mt << 4;
  const int nBase = g << 7;
  const int NdMax = Nd - 1;
  const int kBeg = blockIdx.y * kChunk;
  const int kEnd = (Kd < kBeg + kChunk) ? Kd : (kBeg + kChunk);
  const int kFull = kBeg + ((kEnd - kBeg) & ~31);

  v8f acc = {};
  for (int k0 = kBeg; k0 < kFull; k0 += 32) {
    stage_tiles<false>(A, ldaR, ldaC, B, ldbR, ldbC, As, Bs,
                       m0, nBase, NdMax, k0, kEnd, tid, bNFast);
    if (k0 + 32 < kEnd)   // speculative prefetch of next A row
      __builtin_prefetch(A + (long long)(m0 + (tid & 15)) * ldaR +
                             (long long)(k0 + 32) * ldaC, 0, 1);
    __syncthreads();
    acc = wmma_step(As, Bs, lane, w, acc);
    __syncthreads();
  }
  if (kFull < kEnd) {
    stage_tiles<true>(A, ldaR, ldaC, B, ldbR, ldbC, As, Bs,
                      m0, nBase, NdMax, kFull, kEnd, tid, bNFast);
    __syncthreads();
    acc = wmma_step(As, Bs, lane, w, acc);
  }

  const int ncol0 = nBase + (w << 4);
  if (ncol0 < Nd) {                        // wave-uniform validity
    const int ncol = ncol0 + (lane & 15);
    if (atomicMode) {
#pragma unroll
      for (int gg = 0; gg < 8; ++gg) {
        int r = m0 + gg + ((lane & 16) ? 8 : 0);
        atomicAdd(&C[(long long)r * ldc + ncol], acc[gg]);
      }
    } else {
      float bv = bias ? bias[ncol] : 0.f;
#pragma unroll
      for (int gg = 0; gg < 8; ++gg) {
        int r = m0 + gg + ((lane & 16) ? 8 : 0);
        float v = acc[gg] + bv;
        if (relu) v = fmaxf(v, 0.f);
        C[(long long)r * ldc + ncol] = v;
      }
    }
  }
}

// --------------------------- elementwise / sparse ---------------------------
__global__ void zero_k(float* __restrict__ p, long long n) {
  long long i = (long long)blockIdx.x * blockDim.x + threadIdx.x;
  if (i < n) p[i] = 0.f;
}

__global__ void deg_k(const int* __restrict__ ei, const float* __restrict__ ew,
                      float* __restrict__ deg, int E) {
  int e = blockIdx.x * blockDim.x + threadIdx.x;
  if (e < E) atomicAdd(&deg[ei[e]], ew[e]);
}

__global__ void normw_k(const int* __restrict__ ei, const float* __restrict__ ew,
                        const float* __restrict__ deg, float* __restrict__ nw, int E) {
  int e = blockIdx.x * blockDim.x + threadIdx.x;
  if (e < E) {
    float d = deg[ei[e]];
    d = (d < 0.5f) ? d + 1.f : d;
    nw[e] = ew[e] / d;
  }
}

__global__ void embed_k(const int* __restrict__ x, const float* __restrict__ tbl,
                        float* __restrict__ h, long long tot) {
  long long i = (long long)blockIdx.x * blockDim.x + threadIdx.x;
  if (i < tot) h[i] = tbl[(long long)x[i >> 6] * 64 + (i & 63)];
}

// per-column sum / sumsq: one block per channel
__global__ void colstats(const float* __restrict__ x, long long ld, int n,
                         float* __restrict__ sum, float* __restrict__ sumsq) {
  int c = blockIdx.x;
  float s = 0.f, s2 = 0.f;
  for (int i = threadIdx.x; i < n; i += blockDim.x) {
    float v = x[(long long)i * ld + c];
    s += v; s2 += v * v;
  }
  __shared__ float ls[256], ls2[256];
  ls[threadIdx.x] = s; ls2[threadIdx.x] = s2;
  __syncthreads();
  for (int off = 128; off > 0; off >>= 1) {
    if (threadIdx.x < off) { ls[threadIdx.x] += ls[threadIdx.x + off];
                             ls2[threadIdx.x] += ls2[threadIdx.x + off]; }
    __syncthreads();
  }
  if (threadIdx.x == 0) { sum[c] = ls[0]; sumsq[c] = ls2[0]; }
}

// GraphNorm apply: y = w*(x - ms*mu)/sqrt(var+eps) + b  (optionally relu)
__global__ void gn_apply(const float* __restrict__ xin, long long ldin,
                         float* __restrict__ xout, long long ldout,
                         const float* __restrict__ w, const float* __restrict__ b,
                         const float* __restrict__ ms,
                         const float* __restrict__ sum, const float* __restrict__ sumsq,
                         int n, int C, int relu) {
  long long idx = (long long)blockIdx.x * blockDim.x + threadIdx.x;
  if (idx >= (long long)n * C) return;
  int i = (int)(idx / C), c = (int)(idx % C);
  float invn = 1.f / (float)n;
  float mu = sum[c] * invn, ex2 = sumsq[c] * invn, m = ms[c];
  float var = ex2 - 2.f * m * mu * mu + m * m * mu * mu;
  float v = (xin[(long long)i * ldin + c] - m * mu) * rsqrtf(var + 1e-5f);
  v = w[c] * v + b[c];
  if (relu) v = fmaxf(v, 0.f);
  xout[(long long)i * ldout + c] = v;
}

// SpMM scatter: hcat[row, loff+c] += norm_w[e] * hlin[col, c]
__global__ void spmm_k(const int* __restrict__ ei, const float* __restrict__ nw,
                       const float* __restrict__ hlin, float* __restrict__ hcat,
                       int E, int loff) {
  long long t = (long long)blockIdx.x * blockDim.x + threadIdx.x;
  if (t >= (long long)E * 16) return;
  int e = (int)(t >> 4), c0 = (int)(t & 15) * 4;
  int r = ei[e], cn = ei[E + e];
  float w = nw[e];
  const float4 hv = *(const float4*)(hlin + (long long)cn * 64 + c0);
  float* dst = hcat + (long long)r * F_CAT + loff + c0;
  atomicAdd(dst + 0, w * hv.x);
  atomicAdd(dst + 1, w * hv.y);
  atomicAdd(dst + 2, w * hv.z);
  atomicAdd(dst + 3, w * hv.w);
}

// row softmax over 16 logits (in-place) + column sums + mincut denominator
__global__ void softmax_den(float* __restrict__ s, const float* __restrict__ deg_raw,
                            float* __restrict__ clusum, float* __restrict__ den, int n) {
  __shared__ float lclu[KCLU];
  __shared__ float lden[256];
  if (threadIdx.x < KCLU) lclu[threadIdx.x] = 0.f;
  __syncthreads();
  int i = blockIdx.x * blockDim.x + threadIdx.x;
  float dden = 0.f;
  if (i < n) {
    float v[KCLU]; float mx = -1e30f;
#pragma unroll
    for (int c = 0; c < KCLU; ++c) { v[c] = s[i * KCLU + c]; mx = fmaxf(mx, v[c]); }
    float ssum = 0.f;
#pragma unroll
    for (int c = 0; c < KCLU; ++c) { v[c] = __expf(v[c] - mx); ssum += v[c]; }
    float inv = 1.f / ssum, sq = 0.f;
#pragma unroll
    for (int c = 0; c < KCLU; ++c) {
      v[c] *= inv; s[i * KCLU + c] = v[c]; sq += v[c] * v[c];
      atomicAdd(&lclu[c], v[c]);
    }
    dden = deg_raw[i] * sq;
  }
  lden[threadIdx.x] = dden;
  __syncthreads();
  for (int off = 128; off > 0; off >>= 1) {
    if (threadIdx.x < off) lden[threadIdx.x] += lden[threadIdx.x + off];
    __syncthreads();
  }
  if (threadIdx.x == 0) atomicAdd(den, lden[0]);
  if (threadIdx.x < KCLU) atomicAdd(&clusum[threadIdx.x], lclu[threadIdx.x]);
}

__global__ void mincut_num_k(const int* __restrict__ ei, const float* __restrict__ ew,
                             const float* __restrict__ ssm, float* __restrict__ num, int E) {
  __shared__ float l[256];
  int e = blockIdx.x * blockDim.x + threadIdx.x;
  float a = 0.f;
  if (e < E) {
    int r = ei[e], c = ei[E + e];
    float d = 0.f;
#pragma unroll
    for (int k = 0; k < KCLU; ++k) d += ssm[r * KCLU + k] * ssm[c * KCLU + k];
    a = ew[e] * d;
  }
  l[threadIdx.x] = a;
  __syncthreads();
  for (int off = 128; off > 0; off >>= 1) {
    if (threadIdx.x < off) l[threadIdx.x] += l[threadIdx.x + off];
    __syncthreads();
  }
  if (threadIdx.x == 0) atomicAdd(num, l[0]);
}

// per-subgraph: L1-scale scores, argsort-desc 16 clusters, concat [out|score]
__global__ void pool_k(const float* __restrict__ sub2clu, const float* __restrict__ clusum,
                       const float* __restrict__ outK, float* __restrict__ emb) {
  int m = blockIdx.x, t = threadIdx.x;
  __shared__ float r[KCLU];
  __shared__ int perm[KCLU];
  if (t < KCLU) r[t] = sub2clu[t * M_SUB + m] / fmaxf(clusum[t], 1e-12f);
  __syncthreads();
  if (t < KCLU) {
    float ri = r[t]; int rank = 0;
#pragma unroll
    for (int j = 0; j < KCLU; ++j) {
      float rj = r[j];
      if (rj > ri || (rj == ri && j < t)) rank++;
    }
    perm[rank] = t;
  }
  __syncthreads();
  for (int idx = t; idx < KCLU * (F_CAT + 1); idx += blockDim.x) {
    int p = idx / (F_CAT + 1), c = idx % (F_CAT + 1);
    int cl = perm[p];
    emb[(long long)m * (KCLU * (F_CAT + 1)) + idx] =
        (c < F_CAT) ? outK[cl * F_CAT + c] : r[cl];
  }
}

// mc_loss and o_loss (ss is 16x16, ||ss/||ss||_F - I/4||_F)
__global__ void finalize_k(const float* __restrict__ ss, const float* __restrict__ num,
                           const float* __restrict__ den, float* __restrict__ out) {
  __shared__ float l[256];
  int t = threadIdx.x;
  float v = ss[t];
  l[t] = v * v;
  __syncthreads();
  for (int off = 128; off > 0; off >>= 1) {
    if (t < off) l[t] += l[t + off];
    __syncthreads();
  }
  float nrm = sqrtf(l[0]);
  __syncthreads();
  float e = v / nrm - ((t / 16 == t % 16) ? 0.25f : 0.f);
  l[t] = e * e;
  __syncthreads();
  for (int off = 128; off > 0; off >>= 1) {
    if (t < off) l[t] += l[t + off];
    __syncthreads();
  }
  if (t == 0) {
    out[M_SUB * 64]     = -(num[0] / den[0]);
    out[M_SUB * 64 + 1] = sqrtf(l[0]);
  }
}

// ---------------------------------------------------------------------------
static inline void launch_gemm(const float* A, long long laR, long long laC,
                               const float* B, long long lbR, long long lbC,
                               const float* bias, float* C, long long ldc,
                               int M, int Nd, int Kd, int kChunk, int relu,
                               int atomicMode, hipStream_t stream) {
  int ntg = (Nd / 16 + 7) / 8;                 // 128-wide column groups
  int bNFast = (lbC == 1) ? 1 : 0;             // coalesce along the unit B stride
  dim3 g((unsigned)((M / 16) * ntg), (unsigned)((Kd + kChunk - 1) / kChunk), 1);
  gemm_wmma<<<g, 256, 0, stream>>>(A, laR, laC, B, lbR, lbC, bias, C, ldc,
                                   M, Nd, Kd, kChunk, relu, atomicMode, ntg,
                                   bNFast);
}

extern "C" void kernel_launch(void* const* d_in, const int* in_sizes, int n_in,
                              void* d_out, int out_size, void* d_ws, size_t ws_size,
                              hipStream_t stream) {
  const int    N = N_NODES, E = N_EDGES;
  const int*   x   = (const int*)d_in[0];
  const int*   ei  = (const int*)d_in[1];
  const float* ew  = (const float*)d_in[2];
  const float* sa  = (const float*)d_in[4];          // [M_SUB, N]
  const float* tbl = (const float*)d_in[5];
  const float* egw = (const float*)d_in[6];
  const float* egb = (const float*)d_in[7];
  const float* egm = (const float*)d_in[8];
  const float* cW  = (const float*)d_in[9];          // [3,64,64]
  const float* cb  = (const float*)d_in[10];         // [3,64]
  const float* gw  = (const float*)d_in[11];         // [2,64]
  const float* gb  = (const float*)d_in[12];
  const float* gm  = (const float*)d_in[13];
  const float* glw = (const float*)d_in[14];         // [192]
  const float* glb = (const float*)d_in[15];
  const float* glm = (const float*)d_in[16];
  const float* sW  = (const float*)d_in[17];         // [192,16]
  const float* sb  = (const float*)d_in[18];
  const float* pW1 = (const float*)d_in[19];         // [3088,128]
  const float* pb1 = (const float*)d_in[20];
  const float* pW2 = (const float*)d_in[21];
  const float* pb2 = (const float*)d_in[22];
  const float* pW3 = (const float*)d_in[23];
  const float* pb3 = (const float*)d_in[24];
  const float* pW4 = (const float*)d_in[25];         // [128,64]
  const float* pb4 = (const float*)d_in[26];
  float* out = (float*)d_out;
  float* ws  = (float*)d_ws;

  // workspace layout (floats)
  float* deg    = ws;                         // 50000
  float* nw     = deg    + 50000;             // 800000
  float* h      = nw     + 800000;            // N*64
  float* hlin   = h      + (long long)N * 64; // N*64
  float* hcat   = hlin   + (long long)N * 64; // N*192 (accumulated)
  float* hcat_n = hcat   + (long long)N * F_CAT;
  float* s_buf  = hcat_n + (long long)N * F_CAT;   // N*16
  float* csum   = s_buf  + (long long)N * KCLU;    // 192
  float* csq    = csum   + F_CAT;                  // 192
  float* clusum = csq    + F_CAT;                  // 16
  float* outK   = clusum + KCLU;                   // 16*192
  float* ssm2   = outK   + KCLU * F_CAT;           // 16*16
  float* s2c    = ssm2   + KCLU * KCLU;            // 16*256
  float* scal   = s2c    + KCLU * M_SUB;           // num, den
  float* emb    = scal   + 2;                      // 256*3088
  float* m1     = emb    + (long long)M_SUB * 3088;
  float* m2     = m1     + M_SUB * 128;

  auto Z = [&](float* p, long long n) {
    zero_k<<<dim3((unsigned)((n + 255) / 256)), 256, 0, stream>>>(p, n);
  };
  // zero all accumulation targets (harness does not re-zero between replays)
  Z(deg, 50000);
  Z(hcat, (long long)N * F_CAT);
  Z(clusum, KCLU);
  Z(outK, KCLU * F_CAT);
  Z(ssm2, KCLU * KCLU);
  Z(s2c, KCLU * M_SUB);
  Z(scal, 2);

  // degree + row-normalized edge weights
  deg_k<<<dim3((E + 255) / 256), 256, 0, stream>>>(ei, ew, deg, E);
  normw_k<<<dim3((E + 255) / 256), 256, 0, stream>>>(ei, ew, deg, nw, E);

  // embedding + GraphNorm (in place)
  embed_k<<<dim3((unsigned)(((long long)N * 64 + 255) / 256)), 256, 0, stream>>>(
      x, tbl, h, (long long)N * 64);
  colstats<<<dim3(64), 256, 0, stream>>>(h, 64, N, csum, csq);
  gn_apply<<<dim3((unsigned)(((long long)N * 64 + 255) / 256)), 256, 0, stream>>>(
      h, 64, h, 64, egw, egb, egm, csum, csq, N, 64, 0);

  // 3 GLASSConv layers
  for (int l = 0; l < 3; ++l) {
    launch_gemm(h, 64, 1, cW + (long long)l * 64 * 64, 64, 1, cb + l * 64,
                hlin, 64, N, 64, 64, 64, /*relu=*/1, 0, stream);
    spmm_k<<<dim3((unsigned)(((long long)E * 16 + 255) / 256)), 256, 0, stream>>>(
        ei, nw, hlin, hcat, E, l * 64);
    if (l < 2) {
      colstats<<<dim3(64), 256, 0, stream>>>(hcat + l * 64, F_CAT, N, csum, csq);
      gn_apply<<<dim3((unsigned)(((long long)N * 64 + 255) / 256)), 256, 0, stream>>>(
          hcat + l * 64, F_CAT, h, 64, gw + l * 64, gb + l * 64, gm + l * 64,
          csum, csq, N, 64, /*relu=*/1);
    }
  }

  // final GraphNorm over concat
  colstats<<<dim3(F_CAT), 256, 0, stream>>>(hcat, F_CAT, N, csum, csq);
  gn_apply<<<dim3((unsigned)(((long long)N * F_CAT + 255) / 256)), 256, 0, stream>>>(
      hcat, F_CAT, hcat_n, F_CAT, glw, glb, glm, csum, csq, N, F_CAT, 0);

  // cluster logits + softmax (+ column sums + mincut denominator)
  launch_gemm(hcat_n, F_CAT, 1, sW, KCLU, 1, sb, s_buf, KCLU,
              N, KCLU, F_CAT, F_CAT, 0, 0, stream);
  softmax_den<<<dim3((N + 255) / 256), 256, 0, stream>>>(s_buf, deg, clusum,
                                                         scal + 1, N);
  mincut_num_k<<<dim3((E + 255) / 256), 256, 0, stream>>>(ei, ew, s_buf, scal, E);

  // split-K reductions over all nodes (A = s_sm^T via strides)
  launch_gemm(s_buf, 1, KCLU, hcat_n, F_CAT, 1, nullptr, outK, F_CAT,
              KCLU, F_CAT, N, 800, 0, /*atomic=*/1, stream);            // [16,192]
  launch_gemm(s_buf, 1, KCLU, s_buf, KCLU, 1, nullptr, ssm2, KCLU,
              KCLU, KCLU, N, 800, 0, /*atomic=*/1, stream);             // [16,16]
  launch_gemm(s_buf, 1, KCLU, sa, 1, N_NODES, nullptr, s2c, M_SUB,
              KCLU, M_SUB, N, 800, 0, /*atomic=*/1, stream);            // [16,256]

  // sort-pool per subgraph
  pool_k<<<dim3(M_SUB), 256, 0, stream>>>(s2c, clusum, outK, emb);

  // prediction MLP (all WMMA)
  launch_gemm(emb, 3088, 1, pW1, 128, 1, pb1, m1, 128, M_SUB, 128, 3088, 3088, 1, 0, stream);
  launch_gemm(m1, 128, 1, pW2, 128, 1, pb2, m2, 128, M_SUB, 128, 128, 128, 1, 0, stream);
  launch_gemm(m2, 128, 1, pW3, 128, 1, pb3, m1, 128, M_SUB, 128, 128, 128, 1, 0, stream);
  launch_gemm(m1, 128, 1, pW4, 64, 1, pb4, out, 64, M_SUB, 64, 128, 128, 0, 0, stream);

  // losses
  finalize_k<<<dim3(1), 256, 0, stream>>>(ssm2, scal, scal + 1, out);
}
